// Seq2Queue_55138790146730
// MI455X (gfx1250) — matrix operand order repaired and checked by reading, not compiled, not run on hardware
//
#include <hip/hip_runtime.h>
#include <math.h>

// ---------------------------------------------------------------------------
// Seq2Queue on MI455X (gfx1250, wave32, WMMA 16x16x32 f16 -> f32)
// Pipeline:
//   k_prep_weights : W_ih/W_hh -> f16 (row-major = B-transposed layout), bias sum
//   k_transpose_mem: memory[s][n][c] -> memT16[n][c][s]  (LDS 32x32 tiles)
//   k_actions      : 2-way softmax logits, wave-per-(s,n)
//   k_posvec       : sequential scan, one WG per n, p[256] in LDS,
//                    emits psT16[n][q][s] = posvec * a1  (a1 folded in)
//   k_queue_gemm   : 64 batched [256x512]@[512x512] via v_wmma,
//                    2x2 register-blocked (32x32 per wave) -> queue16
//   k_gates (x256) : gates = bias + x@Wih^T + h@Whh^T via v_wmma; one wave
//                    owns a full 64-row x 16-col strip (weights loaded once)
//   k_cell  (x256) : LSTM cell elementwise, writes hs[t], h16 for next step
//   k_ln_dec       : LayerNorm + [512x10] decode, wave-per-row reductions
// ---------------------------------------------------------------------------

typedef __attribute__((ext_vector_type(16))) _Float16 v16h;
typedef __attribute__((ext_vector_type(8)))  _Float16 v8h;
typedef __attribute__((ext_vector_type(8)))  float    v8f;

#define S_   512
#define N_   64
#define C_   512
#define Q_   256
#define V_   10
#define G4C_ 2048

// ---------------- wave32 helpers ----------------
__device__ __forceinline__ float warp_sum(float v) {
#pragma unroll
  for (int m = 16; m >= 1; m >>= 1) v += __shfl_xor(v, m, 32);
  return v;
}

// A fragment: 16x32 (MxK) f16, row-major source (ld = K-stride).
// ISA layout: lanes 0-15 row M=lane, K={0..7,16..23}; lanes 16-31 K={8..15,24..31}.
__device__ __forceinline__ v16h frag_a(const _Float16* base, int ld, int row0,
                                       int k0, int lane) {
  const _Float16* p =
      base + (size_t)(row0 + (lane & 15)) * ld + k0 + (lane >> 4) * 8;
  v8h lo = *(const v8h*)p;          // K = k0 + hsel*8 + {0..7}
  v8h hi = *(const v8h*)(p + 16);   // K = k0 + 16 + hsel*8 + {0..7}
  v16h a;
#pragma unroll
  for (int e = 0; e < 8; ++e) { a[e] = lo[e]; a[8 + e] = hi[e]; }
  return a;
}

// B fragment: 32x16 (KxN) f16, source stored transposed [N][K] (ld = K-stride).
// Lanes 0-15 col N=lane hold K={0..7,16..23}; lanes 16-31 hold K={8..15,24..31}.
__device__ __forceinline__ v16h frag_b(const _Float16* baseT, int ld, int col0,
                                       int k0, int lane) {
  const _Float16* p =
      baseT + (size_t)(col0 + (lane & 15)) * ld + k0 + (lane >> 4) * 8;
  v8h lo = *(const v8h*)p;
  v8h hi = *(const v8h*)(p + 16);
  v16h b;
#pragma unroll
  for (int e = 0; e < 8; ++e) { b[e] = lo[e]; b[8 + e] = hi[e]; }
  return b;
}

// ---------------- stage kernels ----------------
__global__ void k_prep_weights(const float* __restrict__ Wih,
                               const float* __restrict__ Whh,
                               const float* __restrict__ bih,
                               const float* __restrict__ bhh,
                               _Float16* __restrict__ Wih16,
                               _Float16* __restrict__ Whh16,
                               float* __restrict__ bias) {
  size_t i = (size_t)blockIdx.x * blockDim.x + threadIdx.x;
  if (i < (size_t)G4C_ * C_) {
    Wih16[i] = (_Float16)Wih[i];
    Whh16[i] = (_Float16)Whh[i];
  }
  if (i < G4C_) bias[i] = bih[i] + bhh[i];
}

__global__ void k_transpose_mem(const float* __restrict__ mem,
                                _Float16* __restrict__ memT) {
  __shared__ float tile[32][33];
  int b  = blockIdx.x;
  int n  = b >> 8;
  int st = (b >> 4) & 15;
  int ct = b & 15;
  int tx = threadIdx.x & 31;
  int ty = threadIdx.x >> 5;  // 0..7
#pragma unroll
  for (int i = 0; i < 4; ++i) {
    int s = st * 32 + ty + i * 8;
    int c = ct * 32 + tx;
    tile[ty + i * 8][tx] = mem[((size_t)s * N_ + n) * C_ + c];
  }
  __syncthreads();
#pragma unroll
  for (int i = 0; i < 4; ++i) {
    int c = ct * 32 + ty + i * 8;
    int s = st * 32 + tx;
    memT[((size_t)n * C_ + c) * S_ + s] = (_Float16)tile[tx][ty + i * 8];
  }
}

__global__ void k_actions(const float* __restrict__ mem,
                          const float* __restrict__ W_act,
                          const float* __restrict__ b_act,
                          float* __restrict__ act0, float* __restrict__ act1) {
  int w    = blockIdx.x * 8 + (threadIdx.x >> 5);  // w = s*64 + n
  int lane = threadIdx.x & 31;
  const float* row = mem + (size_t)w * C_;
  float s0 = 0.f, s1 = 0.f;
  for (int c = lane; c < C_; c += 32) {
    float m = row[c];
    s0 += m * W_act[2 * c];
    s1 += m * W_act[2 * c + 1];
  }
  s0 = warp_sum(s0);
  s1 = warp_sum(s1);
  if (lane == 0) {
    float z0 = s0 + b_act[0], z1 = s1 + b_act[1];
    float mx = fmaxf(z0, z1);
    float e0 = expf(z0 - mx), e1 = expf(z1 - mx);
    float inv = 1.f / (e0 + e1);
    act0[w] = e0 * inv;
    act1[w] = e1 * inv;
  }
}

// p_{s+1}[q] = p_s[q]*a0 + p_s[q-1]*a1 ; emit psT[n][q][s] = p_s[q] * a1[s,n]
__global__ void k_posvec(const float* __restrict__ act0,
                         const float* __restrict__ act1,
                         _Float16* __restrict__ psT) {
  __shared__ float p[Q_];
  int n = blockIdx.x;
  int q = threadIdx.x;
  p[q] = (q == 0) ? 1.f : 0.f;
  __syncthreads();
  for (int s = 0; s < S_; ++s) {
    float a0 = act0[s * N_ + n];
    float a1 = act1[s * N_ + n];
    float pc = p[q];
    float pm = p[(q + Q_ - 1) & (Q_ - 1)];
    psT[((size_t)n * Q_ + q) * S_ + s] = (_Float16)(pc * a1);
    __syncthreads();
    p[q] = pc * a0 + pm * a1;
    __syncthreads();
  }
}

// queue[q][n][c] = sum_s psT[n][q][s] * memT[n][c][s]   (64 batched GEMMs)
// 2x2 register blocking: each wave computes a 32x32 output tile.
__global__ void k_queue_gemm(const _Float16* __restrict__ psT,
                             const _Float16* __restrict__ memT,
                             _Float16* __restrict__ queue16) {
  int w    = blockIdx.x * 8 + (threadIdx.x >> 5);
  int lane = threadIdx.x & 31;
  int n   = w >> 7;         // 128 macro-tiles per batch
  int rem = w & 127;
  int q0  = (rem >> 4) * 32;  // 8 q macro-tiles
  int c0  = (rem & 15) * 32;  // 16 c macro-tiles
  const _Float16* A = psT + (size_t)n * Q_ * S_;
  const _Float16* B = memT + (size_t)n * C_ * S_;
  v8f acc00 = {}, acc01 = {}, acc10 = {}, acc11 = {};
  for (int k0 = 0; k0 < S_; k0 += 32) {
    __builtin_prefetch(
        (const void*)(B + (size_t)(c0 + (lane & 15)) * S_ + k0 + 32), 0, 3);
    v16h a0 = frag_a(A, S_, q0, k0, lane);
    v16h a1 = frag_a(A, S_, q0 + 16, k0, lane);
    v16h b0 = frag_b(B, S_, c0, k0, lane);
    v16h b1 = frag_b(B, S_, c0 + 16, k0, lane);
    acc00 = __builtin_amdgcn_wmma_f32_16x16x32_f16(false, a0, false, b0,
                                                   (short)0, acc00, false, false);
    acc01 = __builtin_amdgcn_wmma_f32_16x16x32_f16(false, a0, false, b1,
                                                   (short)0, acc01, false, false);
    acc10 = __builtin_amdgcn_wmma_f32_16x16x32_f16(false, a1, false, b0,
                                                   (short)0, acc10, false, false);
    acc11 = __builtin_amdgcn_wmma_f32_16x16x32_f16(false, a1, false, b1,
                                                   (short)0, acc11, false, false);
  }
  int col = lane & 15;
  int r0  = 8 * (lane >> 4);
#pragma unroll
  for (int r = 0; r < 8; ++r) {
    int qa = q0 + r0 + r, qb = q0 + 16 + r0 + r;
    queue16[((size_t)qa * N_ + n) * C_ + c0 + col]      = (_Float16)acc00[r];
    queue16[((size_t)qa * N_ + n) * C_ + c0 + 16 + col] = (_Float16)acc01[r];
    queue16[((size_t)qb * N_ + n) * C_ + c0 + col]      = (_Float16)acc10[r];
    queue16[((size_t)qb * N_ + n) * C_ + c0 + 16 + col] = (_Float16)acc11[r];
  }
}

__global__ void k_init_state(_Float16* __restrict__ h16,
                             float* __restrict__ cst) {
  int i = blockIdx.x * blockDim.x + threadIdx.x;
  if (i < N_ * C_) {
    h16[i] = (_Float16)0.f;
    cst[i] = 0.f;
  }
}

// gates[n][j] = bias[j] + sum_c x[n][c]*Wih[j][c] + sum_c h[n][c]*Whh[j][c]
// One wave owns all 64 rows (4 M-tiles) of a 16-column weight strip, so each
// weight fragment is fetched exactly once per step (4 MB/step minimum).
__global__ void k_gates(const _Float16* __restrict__ queue16,
                        const _Float16* __restrict__ h16,
                        const _Float16* __restrict__ Wih16,
                        const _Float16* __restrict__ Whh16,
                        const float* __restrict__ bias,
                        float* __restrict__ gates, int t) {
  int jt   = blockIdx.x * 8 + (threadIdx.x >> 5);  // 128 col-tiles (4C=2048)
  int lane = threadIdx.x & 31;
  const _Float16* X = queue16 + (size_t)t * N_ * C_;
  float bv = bias[jt * 16 + (lane & 15)];
  v8f acc[4];
#pragma unroll
  for (int nt = 0; nt < 4; ++nt)
#pragma unroll
    for (int r = 0; r < 8; ++r) acc[nt][r] = bv;
  for (int k0 = 0; k0 < C_; k0 += 32) {
    __builtin_prefetch(
        (const void*)(Whh16 + (size_t)(jt * 16 + (lane & 15)) * C_ + k0 + 32),
        0, 3);
    v16h bi = frag_b(Wih16, C_, jt * 16, k0, lane);
    v16h bh = frag_b(Whh16, C_, jt * 16, k0, lane);
#pragma unroll
    for (int nt = 0; nt < 4; ++nt) {
      v16h ax = frag_a(X, C_, nt * 16, k0, lane);
      acc[nt] = __builtin_amdgcn_wmma_f32_16x16x32_f16(false, ax, false, bi,
                                                       (short)0, acc[nt],
                                                       false, false);
      v16h ah = frag_a(h16, C_, nt * 16, k0, lane);
      acc[nt] = __builtin_amdgcn_wmma_f32_16x16x32_f16(false, ah, false, bh,
                                                       (short)0, acc[nt],
                                                       false, false);
    }
  }
  int col = jt * 16 + (lane & 15);
#pragma unroll
  for (int nt = 0; nt < 4; ++nt) {
    int r0 = nt * 16 + 8 * (lane >> 4);
#pragma unroll
    for (int r = 0; r < 8; ++r)
      gates[(size_t)(r0 + r) * G4C_ + col] = acc[nt][r];
  }
}

__global__ void k_cell(const float* __restrict__ gates, float* __restrict__ cst,
                       float* __restrict__ hs, _Float16* __restrict__ h16,
                       int t) {
  int i = blockIdx.x * blockDim.x + threadIdx.x;  // i = n*512 + c
  int n = i >> 9, c = i & 511;
  const float* g = gates + (size_t)n * G4C_;
  float gi = g[c], gf = g[C_ + c], gg = g[2 * C_ + c], go = g[3 * C_ + c];
  float si = 1.f / (1.f + expf(-gi));
  float sf = 1.f / (1.f + expf(-gf));
  float so = 1.f / (1.f + expf(-go));
  float cs = sf * cst[i] + si * tanhf(gg);
  float h  = so * tanhf(cs);
  cst[i] = cs;
  hs[(size_t)t * (N_ * C_) + i] = h;
  h16[i] = (_Float16)h;
}

__global__ void k_ln_dec(const float* __restrict__ hs,
                         const float* __restrict__ gamma,
                         const float* __restrict__ beta,
                         const float* __restrict__ Wdec,
                         const float* __restrict__ bdec,
                         float* __restrict__ out) {
  int row  = blockIdx.x * 8 + (threadIdx.x >> 5);  // row = q*64 + n
  int lane = threadIdx.x & 31;
  const float* x = hs + (size_t)row * C_;
  float nv[16];
  float s1 = 0.f, s2 = 0.f;
#pragma unroll
  for (int k = 0; k < 16; ++k) {
    float v = x[lane + 32 * k];
    nv[k] = v;
    s1 += v;
    s2 += v * v;
  }
  s1 = warp_sum(s1);
  s2 = warp_sum(s2);
  float mean = s1 * (1.f / C_);
  float var  = s2 * (1.f / C_) - mean * mean;
  float rstd = rsqrtf(var + 1e-5f);
#pragma unroll
  for (int k = 0; k < 16; ++k) {
    int c = lane + 32 * k;
    nv[k] = (nv[k] - mean) * rstd * gamma[c] + beta[c];
  }
  for (int v = 0; v < V_; ++v) {
    float acc = 0.f;
#pragma unroll
    for (int k = 0; k < 16; ++k) {
      int c = lane + 32 * k;
      acc += nv[k] * Wdec[c * V_ + v];
    }
    acc = warp_sum(acc);
    if (lane == 0) out[(size_t)row * V_ + v] = acc + bdec[v];
  }
}

// ---------------- host launcher ----------------
extern "C" void kernel_launch(void* const* d_in, const int* in_sizes, int n_in,
                              void* d_out, int out_size, void* d_ws,
                              size_t ws_size, hipStream_t stream) {
  (void)in_sizes; (void)n_in; (void)out_size; (void)ws_size;
  const float* memory = (const float*)d_in[0];
  const float* W_act  = (const float*)d_in[1];
  const float* b_act  = (const float*)d_in[2];
  const float* W_ih   = (const float*)d_in[3];
  const float* W_hh   = (const float*)d_in[4];
  const float* b_ih   = (const float*)d_in[5];
  const float* b_hh   = (const float*)d_in[6];
  const float* ln_g   = (const float*)d_in[7];
  const float* ln_b   = (const float*)d_in[8];
  const float* W_dec  = (const float*)d_in[9];
  const float* b_dec  = (const float*)d_in[10];
  float* out = (float*)d_out;

  char* ws   = (char*)d_ws;
  size_t off = 0;
  auto alloc = [&](size_t bytes) -> char* {
    char* p = ws + off;
    off += (bytes + 255) & ~(size_t)255;
    return p;
  };
  float*    act0    = (float*)alloc((size_t)S_ * N_ * 4);
  float*    act1    = (float*)alloc((size_t)S_ * N_ * 4);
  float*    bias    = (float*)alloc((size_t)G4C_ * 4);
  _Float16* Wih16   = (_Float16*)alloc((size_t)G4C_ * C_ * 2);
  _Float16* Whh16   = (_Float16*)alloc((size_t)G4C_ * C_ * 2);
  _Float16* memT    = (_Float16*)alloc((size_t)N_ * C_ * S_ * 2);
  _Float16* psT     = (_Float16*)alloc((size_t)N_ * Q_ * S_ * 2);
  _Float16* queue16 = (_Float16*)alloc((size_t)Q_ * N_ * C_ * 2);
  float*    gates   = (float*)alloc((size_t)N_ * G4C_ * 4);
  _Float16* h16     = (_Float16*)alloc((size_t)N_ * C_ * 2);
  float*    cst     = (float*)alloc((size_t)N_ * C_ * 4);
  float*    hs      = (float*)alloc((size_t)Q_ * N_ * C_ * 4);

  k_prep_weights<<<4096, 256, 0, stream>>>(W_ih, W_hh, b_ih, b_hh, Wih16,
                                           Whh16, bias);
  k_transpose_mem<<<16384, 256, 0, stream>>>(memory, memT);
  k_actions<<<4096, 256, 0, stream>>>(memory, W_act, b_act, act0, act1);
  k_posvec<<<64, 256, 0, stream>>>(act0, act1, psT);
  k_queue_gemm<<<1024, 256, 0, stream>>>(psT, memT, queue16);
  k_init_state<<<128, 256, 0, stream>>>(h16, cst);
  for (int t = 0; t < Q_; ++t) {
    k_gates<<<16, 256, 0, stream>>>(queue16, h16, Wih16, Whh16, bias, gates, t);
    k_cell<<<128, 256, 0, stream>>>(gates, cst, hs, h16, t);
  }
  k_ln_dec<<<2048, 256, 0, stream>>>(hs, ln_g, ln_b, W_dec, b_dec, out);
}
